// HyenaFilter_50448685859025
// MI455X (gfx1250) — matrix operand order repaired and checked by reading, not compiled, not run on hardware
//
#include <hip/hip_runtime.h>

// ---------------------------------------------------------------------------
// Hyena filter + FFT long convolution for MI455X (gfx1250, wave32).
//  - Filter projection GEMM (8192x64 @ 64x768) uses v_wmma_f32_16x16x32_f16.
//  - 16384-pt real FFTs done as 8192-pt complex FFTs fully resident in LDS
//    (64 KB float2 buffer), fused fwd-FFT * K_f -> inv-FFT -> +skip per WG.
//  - Twiddles from an L2-resident master table (no per-butterfly sincos).
//  - Stages with len<=1024 are wave-local (no s_barrier, wave32 lock-step).
// ---------------------------------------------------------------------------

typedef __attribute__((ext_vector_type(16))) _Float16 v16h;
typedef __attribute__((ext_vector_type(8)))  float    v8f;

namespace {
constexpr int SEQ       = 8192;   // sequence length L
constexpr int FFTN      = 8192;   // packed complex FFT size (= 16384 / 2)
constexpr int DM        = 768;    // d_model
constexpr int ORD       = 64;     // filter MLP width
constexpr int EMB       = 33;     // positional embedding dim
constexpr int KF_STRIDE = 8200;   // padded complex stride for 8193 rfft bins
constexpr int THREADS   = 256;    // 8 waves per workgroup
}

// ---------------------------------------------------------------------------
// Kernel 0: master twiddle table W[j] = e^{-2*pi*i*j/FFTN}, j in [0, FFTN/2).
// Stage (len,pos) twiddle = W[pos * (FFTN/len)]; inverse FFT conjugates.
// 32 KB total -> permanently L2-resident across all FFT workgroups.
// ---------------------------------------------------------------------------
__global__ __launch_bounds__(256) void twiddle_init_kernel(float2* __restrict__ tw) {
  const int j = blockIdx.x * blockDim.x + threadIdx.x;
  if (j < FFTN / 2) {
    float s, c;
    sincosf(6.2831853071795864769f * (float)j / (float)FFTN, &s, &c);
    tw[j] = make_float2(c, -s);
  }
}

// ---------------------------------------------------------------------------
// In-place radix-2 DIT FFT on an LDS buffer of 8192 complex floats.
// dir = +1: forward (table holds e^{-i...}), dir = -1: inverse (conjugate).
// Unnormalized; caller scales inverse by 1/N.
// ---------------------------------------------------------------------------
__device__ __forceinline__ void fft8192_lds(float2* buf, const float2* __restrict__ tw,
                                            int tid, float dir) {
  __syncthreads();
  // bit-reversal permutation (13 bits); pairs are disjoint -> no hazards
  for (int i = tid; i < FFTN; i += THREADS) {
    int j = (int)(__brev((unsigned)i) >> 19);
    if (j > i) { float2 a = buf[i]; buf[i] = buf[j]; buf[j] = a; }
  }
  __syncthreads();

  const int wave = tid >> 5;
  const int lane = tid & 31;

  // Wave-local stages: len = 2..1024. Each wave owns an aligned 1024-element
  // chunk (butterflies b in [wave*512, wave*512+512)), so no cross-wave
  // communication; wave32 lanes are lock-step and per-wave DS ops in-order.
  for (int len = 2; len <= 1024; len <<= 1) {
    const int half = len >> 1;
    const int step = (FFTN / 2) / half;
#pragma unroll 1
    for (int it = 0; it < 512 / 32; ++it) {
      const int b = wave * 512 + it * 32 + lane;
      const int pos = b & (half - 1);
      const int i = 2 * b - pos;        // = (b/half)*len + pos
      const int j = i + half;
      const float2 w = tw[pos * step];
      const float wr = w.x, wi = dir * w.y;
      float2 u = buf[i], v = buf[j];
      const float tr = wr * v.x - wi * v.y;
      const float ti = wr * v.y + wi * v.x;
      buf[i] = make_float2(u.x + tr, u.y + ti);
      buf[j] = make_float2(u.x - tr, u.y - ti);
    }
    __builtin_amdgcn_wave_barrier();    // compile-time fence between stages
  }
  __syncthreads();

  // Cross-wave stages: len = 2048, 4096, 8192.
  for (int len = 2048; len <= FFTN; len <<= 1) {
    const int half = len >> 1;
    const int step = (FFTN / 2) / half;
    for (int b = tid; b < FFTN / 2; b += THREADS) {
      const int pos = b & (half - 1);
      const int i = 2 * b - pos;
      const int j = i + half;
      const float2 w = tw[pos * step];
      const float wr = w.x, wi = dir * w.y;
      float2 u = buf[i], v = buf[j];
      const float tr = wr * v.x - wi * v.y;
      const float ti = wr * v.y + wi * v.x;
      buf[i] = make_float2(u.x + tr, u.y + ti);
      buf[j] = make_float2(u.x - tr, u.y - ti);
    }
    __syncthreads();
  }
}

// ---------------------------------------------------------------------------
// Kernel 1: positional-embedding MLP -> H3 (8192 x 64), stored as f16 for WMMA.
// One thread per sequence position; fully unrolled FMA chains.
// ---------------------------------------------------------------------------
__global__ __launch_bounds__(256) void filter_mlp_kernel(
    const float* __restrict__ z, const float* __restrict__ freq,
    const float* __restrict__ W1, const float* __restrict__ b1,
    const float* __restrict__ W2, const float* __restrict__ b2,
    const float* __restrict__ W3, const float* __restrict__ b3,
    _Float16* __restrict__ h3out) {
  const int l = blockIdx.x * blockDim.x + threadIdx.x;
  if (l >= SEQ) return;

  float zi[EMB];
#pragma unroll
  for (int e = 0; e < EMB; ++e) zi[e] = z[l * EMB + e];

  float h1[ORD];
#pragma unroll
  for (int o = 0; o < ORD; ++o) {
    float acc = b1[o];
#pragma unroll
    for (int e = 0; e < EMB; ++e) acc = fmaf(zi[e], W1[o * EMB + e], acc);
    h1[o] = __sinf(freq[o] * acc);
  }

  float h2[ORD];
#pragma unroll
  for (int p = 0; p < ORD; ++p) {
    float acc = b2[p];
#pragma unroll
    for (int o = 0; o < ORD; ++o) acc = fmaf(h1[o], W2[p * ORD + o], acc);
    h2[p] = __sinf(freq[p] * acc);
  }

#pragma unroll
  for (int p = 0; p < ORD; ++p) {
    float acc = b3[p];
#pragma unroll
    for (int o = 0; o < ORD; ++o) acc = fmaf(h2[o], W3[p * ORD + o], acc);
    h3out[(size_t)l * ORD + p] = (_Float16)__sinf(freq[p] * acc);
  }
}

// ---------------------------------------------------------------------------
// Kernel 2: K = H3 (8192x64,f16) @ Wout^T (64x768) via v_wmma_f32_16x16x32_f16,
// epilogue applies decay exp(-t[l]*|delta[d]|) and stores transposed k[d][l].
// One wave per 16x16 tile; two WMMA ops cover K=64.
// ---------------------------------------------------------------------------
__global__ __launch_bounds__(256) void wout_wmma_kernel(
    const _Float16* __restrict__ h3, const float* __restrict__ Wout,
    const float* __restrict__ t, const float* __restrict__ deltas,
    float* __restrict__ kbuf) {
  const int wave = threadIdx.x >> 5;
  const int lane = threadIdx.x & 31;
  const int tile = blockIdx.x * (blockDim.x >> 5) + wave;
  const int DT = DM / 16;                 // 48 tiles along d
  const int tl = tile / DT;
  const int td = tile % DT;
  if (tl >= SEQ / 16) return;             // uniform per wave
  const int L0 = tl * 16;
  const int D0 = td * 16;

  const int row = lane & 15;              // A: M index
  const int n   = lane & 15;              // B/D: N index
  const int kbaseA = (lane < 16) ? 0 : 8; // A K-group per ISA layout
  const int kbaseB = (lane < 16) ? 0 : 16;// unused; B uses kbaseA (same k order as A)

  v8f c = {};
#pragma unroll
  for (int k0 = 0; k0 < ORD; k0 += 32) {
    v16h a, b;
#pragma unroll
    for (int h = 0; h < 16; ++h) {
      const int j = h >> 1, s = h & 1;
      const int ka = (j < 4) ? (kbaseA + 2 * j + s)
                             : (16 + kbaseA + 2 * (j - 4) + s);
      a[h] = h3[(size_t)(L0 + row) * ORD + k0 + ka];
      const int kb = (j < 4) ? (kbaseA + 2 * j + s)
                             : (16 + kbaseA + 2 * (j - 4) + s);
      b[h] = (_Float16)Wout[(size_t)(D0 + n) * ORD + k0 + kb];
    }
    c = __builtin_amdgcn_wmma_f32_16x16x32_f16(
        /*neg_a=*/false, a, /*neg_b=*/false, b,
        /*c_mod=*/(short)0, c, /*reuse_a=*/false, /*reuse_b=*/false);
  }

  const int mbase = (lane < 16) ? 0 : 8;  // C/D layout: VGPR r -> M = r(+8)
#pragma unroll
  for (int r = 0; r < 8; ++r) {
    const int l = L0 + mbase + r;
    const int d = D0 + n;
    const float decay = __expf(-t[l] * fabsf(deltas[d]));
    kbuf[(size_t)d * SEQ + l] = c[r] * decay;
  }
}

// ---------------------------------------------------------------------------
// Kernel 3: filter spectrum K_f[d][0..8192] = rfft(k[d], 16384) via packed
// 8192-pt complex FFT. One workgroup per channel d.
// ---------------------------------------------------------------------------
__global__ __launch_bounds__(256) void kfilter_fft_kernel(
    const float* __restrict__ kbuf, const float2* __restrict__ tw,
    float2* __restrict__ kf) {
  __shared__ float2 sbuf[FFTN];
  const int d = blockIdx.x;
  const int tid = threadIdx.x;
  const float* kd = kbuf + (size_t)d * SEQ;

  for (int nn = tid; nn < FFTN; nn += THREADS) {
    float2 v = make_float2(0.f, 0.f);
    if (nn < SEQ / 2) { v.x = kd[2 * nn]; v.y = kd[2 * nn + 1]; }
    sbuf[nn] = v;
  }
  fft8192_lds(sbuf, tw, tid, +1.f);

  float2* out = kf + (size_t)d * KF_STRIDE;
  const float PIN = 3.14159265358979323846f / (float)FFTN;
  for (int k = tid; k <= FFTN / 2; k += THREADS) {
    if (k == 0) {
      float2 z0 = sbuf[0];
      out[0]    = make_float2(z0.x + z0.y, 0.f);   // U[0]
      out[FFTN] = make_float2(z0.x - z0.y, 0.f);   // U[N]
    } else {
      float2 Za = sbuf[k], Zb = sbuf[FFTN - k];
      float sw, cw;
      sincosf(PIN * (float)k, &sw, &cw);
      const float Ar = Za.x + Zb.x, Ai = Za.y - Zb.y;   // Z[k] + conj(Z[N-k])
      const float Br = Za.x - Zb.x, Bi = Za.y + Zb.y;   // Z[k] - conj(Z[N-k])
      const float p = cw * Br + sw * Bi;                // Re(w*B), w=e^{-i pi k/N}
      const float q = cw * Bi - sw * Br;                // Im(w*B)
      out[k]        = make_float2(0.5f * (Ar + q), 0.5f * (Ai - p));  // U[k]
      out[FFTN - k] = make_float2(0.5f * (Ar - q), 0.5f * (-Ai - p)); // U[N-k]
    }
  }
}

// ---------------------------------------------------------------------------
// Kernel 4: fused y = irfft(rfft(x)*K_f)[:L] + x*bias_D for one (b,d) signal.
// ---------------------------------------------------------------------------
__global__ __launch_bounds__(256) void conv_fft_kernel(
    const float* __restrict__ x, const float2* __restrict__ kf,
    const float2* __restrict__ tw, const float* __restrict__ bias_D,
    float* __restrict__ out) {
  __shared__ float2 sbuf[FFTN];
  const int bd = blockIdx.x;            // b*768 + d
  const int d = bd % DM;
  const int tid = threadIdx.x;
  const float* xp = x + (size_t)bd * SEQ;
  const float2* kfd = kf + (size_t)d * KF_STRIDE;

  // Prefetch the 64 KB filter-spectrum row into cache (global_prefetch_b8);
  // it streams in behind the forward FFT's compute.
  for (int c = tid; c * 16 <= FFTN; c += THREADS)
    __builtin_prefetch((const void*)(kfd + c * 16), 0, 1);

  for (int nn = tid; nn < FFTN; nn += THREADS) {
    float2 v = make_float2(0.f, 0.f);
    if (nn < SEQ / 2) { v.x = xp[2 * nn]; v.y = xp[2 * nn + 1]; }
    sbuf[nn] = v;
  }
  fft8192_lds(sbuf, tw, tid, +1.f);

  // Spectral multiply: unpack U[k],U[N-k] -> P=U*K -> repack for inverse.
  // Each (k, N-k) pair is owned by exactly one thread -> no extra barriers.
  const float PIN = 3.14159265358979323846f / (float)FFTN;
  for (int k = tid; k <= FFTN / 2; k += THREADS) {
    if (k == 0) {
      float2 z0 = sbuf[0];
      const float U0 = z0.x + z0.y, UN = z0.x - z0.y;       // both real
      const float P0 = U0 * kfd[0].x;
      const float PN = UN * kfd[FFTN].x;
      sbuf[0] = make_float2(0.5f * (P0 + PN), 0.5f * (P0 - PN));
    } else {
      float2 Za = sbuf[k], Zb = sbuf[FFTN - k];
      float sw, cw;
      sincosf(PIN * (float)k, &sw, &cw);
      const float Ar = Za.x + Zb.x, Ai = Za.y - Zb.y;
      const float Br = Za.x - Zb.x, Bi = Za.y + Zb.y;
      const float p = cw * Br + sw * Bi;
      const float q = cw * Bi - sw * Br;
      const float2 Uk = make_float2(0.5f * (Ar + q), 0.5f * (Ai - p));
      const float2 Um = make_float2(0.5f * (Ar - q), 0.5f * (-Ai - p));
      const float2 Kk = kfd[k], Km = kfd[FFTN - k];
      const float2 Pk = make_float2(Uk.x * Kk.x - Uk.y * Kk.y,
                                    Uk.x * Kk.y + Uk.y * Kk.x);
      const float2 Pm = make_float2(Um.x * Km.x - Um.y * Km.y,
                                    Um.x * Km.y + Um.y * Km.x);
      // repack: Z'[k] = (Pk+conj(Pm))/2 + i*v*(Pk-conj(Pm))/2, v=e^{+i pi k/N}
      const float A2r = Pk.x + Pm.x, A2i = Pk.y - Pm.y;
      const float B2r = Pk.x - Pm.x, B2i = Pk.y + Pm.y;
      const float pp  = cw * B2r - sw * B2i;   // Re(v*B2)
      const float qq  = cw * B2i + sw * B2r;   // Im(v*B2)
      const float qq2 = -cw * B2i - sw * B2r;  // Im(v2*B2'), v2=(-cw,sw)
      sbuf[k]        = make_float2(0.5f * (A2r - qq),  0.5f * (A2i + pp));
      sbuf[FFTN - k] = make_float2(0.5f * (A2r - qq2), 0.5f * (-A2i + pp));
    }
  }
  fft8192_lds(sbuf, tw, tid, -1.f);

  const float bb = bias_D[d];
  float* op = out + (size_t)bd * SEQ;
  const float inv = 1.0f / (float)FFTN;
  for (int nn = tid; nn < SEQ / 2; nn += THREADS) {
    float2 zv = sbuf[nn];
    op[2 * nn]     = zv.x * inv + xp[2 * nn]     * bb;
    op[2 * nn + 1] = zv.y * inv + xp[2 * nn + 1] * bb;
  }
}

// ---------------------------------------------------------------------------
// Launcher. Input order: x, L, z, t, freq, W1, b1, W2, b2, W3, b3, Wout,
// deltas, bias_D.  Workspace: [twiddles | K_f spectra | k(d,l) | H3 f16].
// ---------------------------------------------------------------------------
extern "C" void kernel_launch(void* const* d_in, const int* in_sizes, int n_in,
                              void* d_out, int out_size, void* d_ws, size_t ws_size,
                              hipStream_t stream) {
  (void)in_sizes; (void)n_in; (void)out_size; (void)ws_size;
  const float* x      = (const float*)d_in[0];
  const float* z      = (const float*)d_in[2];
  const float* t      = (const float*)d_in[3];
  const float* freq   = (const float*)d_in[4];
  const float* W1     = (const float*)d_in[5];
  const float* b1     = (const float*)d_in[6];
  const float* W2     = (const float*)d_in[7];
  const float* b2     = (const float*)d_in[8];
  const float* W3     = (const float*)d_in[9];
  const float* b3     = (const float*)d_in[10];
  const float* Wout   = (const float*)d_in[11];
  const float* deltas = (const float*)d_in[12];
  const float* bias_D = (const float*)d_in[13];
  float* out = (float*)d_out;

  char* ws = (char*)d_ws;
  float2*   tw   = (float2*)ws;                                       // 4096*8 B
  float2*   kf   = (float2*)(ws + (size_t)(FFTN / 2) * sizeof(float2));// 768*8200*8 B
  float*    kbuf = (float*)((char*)kf + (size_t)DM * KF_STRIDE * sizeof(float2));
  _Float16* h3   = (_Float16*)((char*)kbuf + (size_t)DM * SEQ * sizeof(float));

  twiddle_init_kernel<<<(FFTN / 2) / THREADS, THREADS, 0, stream>>>(tw);

  filter_mlp_kernel<<<SEQ / THREADS, THREADS, 0, stream>>>(
      z, freq, W1, b1, W2, b2, W3, b3, h3);

  const int tiles = (SEQ / 16) * (DM / 16);        // 24576 tiles, 8 waves/block
  wout_wmma_kernel<<<tiles / 8, THREADS, 0, stream>>>(h3, Wout, t, deltas, kbuf);

  kfilter_fft_kernel<<<DM, THREADS, 0, stream>>>(kbuf, tw, kf);

  conv_fft_kernel<<<2 * DM, THREADS, 0, stream>>>(x, kf, tw, bias_D, out);
}
